// DoubleStreamBlock_78254304133756
// MI455X (gfx1250) — matrix operand order repaired and checked by reading, not compiled, not run on hardware
//
#include <hip/hip_runtime.h>

// ---------------------------------------------------------------------------
// Types for CDNA5 WMMA (gfx1250, wave32)
// ---------------------------------------------------------------------------
typedef __attribute__((ext_vector_type(16))) __bf16 v16bf;
typedef __attribute__((ext_vector_type(8)))  float  v8f;

union FragB {
    v16bf v;
    unsigned int u[8];
};

// float -> bf16 (round to nearest even), as raw ushort
__device__ __forceinline__ unsigned short f2bf(float x) {
    unsigned u = __float_as_uint(x);
    unsigned r = (u + 0x7FFFu + ((u >> 16) & 1u)) >> 16;
    return (unsigned short)r;
}
__device__ __forceinline__ unsigned pk2bf(float lo, float hi) {
    return (unsigned)f2bf(lo) | ((unsigned)f2bf(hi) << 16);
}

__device__ __forceinline__ v8f wmma_bf16(const FragB& a, const FragB& b, v8f c) {
    return __builtin_amdgcn_wmma_f32_16x16x32_bf16(
        false, a.v, false, b.v, (short)0, c, false, false);
}

// ---------------------------------------------------------------------------
// Async global->LDS staging (CDNA5). Fallback: plain 16B copies.
// Builtin signature (from hipcc diagnostic): param0 = int4* in global AS.
// ---------------------------------------------------------------------------
#if defined(__has_builtin)
#if __has_builtin(__builtin_amdgcn_global_load_async_to_lds_b128)
#define HAVE_ASYNC_LDS 1
#endif
#endif

typedef int v4i __attribute__((vector_size(16)));
typedef __attribute__((address_space(1))) v4i v4i_g;
typedef __attribute__((address_space(3))) v4i v4i_l;

__device__ __forceinline__ void async_copy16(void* lds, const void* g) {
#ifdef HAVE_ASYNC_LDS
    __builtin_amdgcn_global_load_async_to_lds_b128(
        (v4i_g*)g, (v4i_l*)lds, 0, 0);
#else
    *(uint4*)lds = *(const uint4*)g;
#endif
}

#ifdef HAVE_ASYNC_LDS
#define WAIT_ASYNC(n) asm volatile("s_wait_asynccnt " #n ::: "memory")
#else
#define WAIT_ASYNC(n)
#endif

// ---------------------------------------------------------------------------
// Model constants
// ---------------------------------------------------------------------------
#define HS    2048
#define NH    16
#define HD    128
#define LI    2048
#define LT    512
#define LTOT  2560
#define MLPD  8192
#define QKVD  6144
#define MODD  12288

// ---------------------------------------------------------------------------
// Kernel 0: f32 -> bf16 weight conversion (memory bound, one pass per matrix)
// grid (n/2048) x 256, each thread converts 8 elements
// ---------------------------------------------------------------------------
__global__ __launch_bounds__(256) void cvt_bf16_kernel(
    const float* __restrict__ src, unsigned short* __restrict__ dst)
{
    size_t i = ((size_t)blockIdx.x * 256 + threadIdx.x) * 8;
    float4 a = *(const float4*)(src + i);
    float4 b = *(const float4*)(src + i + 4);
    uint4 o;
    o.x = pk2bf(a.x, a.y);
    o.y = pk2bf(a.z, a.w);
    o.z = pk2bf(b.x, b.y);
    o.w = pk2bf(b.z, b.w);
    *(uint4*)(dst + i) = o;
}

// ---------------------------------------------------------------------------
// Kernel 1: mod vectors  m = silu(vec) @ mod_w + mod_b   (1x2048 @ 2048x12288)
// grid (48, 2) x 256
// ---------------------------------------------------------------------------
__global__ __launch_bounds__(256) void mod_kernel(
    const float* __restrict__ vec,
    const float* __restrict__ wI, const float* __restrict__ bI,
    const float* __restrict__ wT, const float* __restrict__ bT,
    float* __restrict__ mI, float* __restrict__ mT)
{
    __shared__ float sv[HS];
    int tid = threadIdx.x;
    for (int i = tid; i < HS; i += 256) {
        float v = vec[i];
        sv[i] = v / (1.f + __expf(-v));
    }
    __syncthreads();
    const float* w; const float* b; float* out;
    if (blockIdx.y == 0) { w = wI; b = bI; out = mI; }
    else                 { w = wT; b = bT; out = mT; }
    int col = blockIdx.x * 256 + tid;
    float acc = 0.f;
    for (int i = 0; i < HS; i++)
        acc += sv[i] * w[(size_t)i * MODD + col];
    out[col] = acc + b[col];
}

// ---------------------------------------------------------------------------
// Kernel 2: LayerNorm + modulation -> bf16
// ---------------------------------------------------------------------------
__global__ __launch_bounds__(256) void ln_mod_kernel(
    const float* __restrict__ x, const float* __restrict__ s,
    const float* __restrict__ b, const float* __restrict__ shift,
    const float* __restrict__ scale, unsigned short* __restrict__ out)
{
    const int C = HS;
    int row = blockIdx.x, tid = threadIdx.x;
    const float* xr = x + (size_t)row * C;
    float lsum = 0.f, lsq = 0.f;
    for (int c = tid; c < C; c += 256) {
        float v = xr[c];
        lsum += v; lsq += v * v;
    }
    for (int off = 16; off > 0; off >>= 1) {
        lsum += __shfl_xor(lsum, off, 32);
        lsq  += __shfl_xor(lsq,  off, 32);
    }
    __shared__ float s1[8], s2[8];
    int wid = tid >> 5, lane = tid & 31;
    if (lane == 0) { s1[wid] = lsum; s2[wid] = lsq; }
    __syncthreads();
    if (tid == 0) {
        float a = 0.f, q = 0.f;
        for (int i = 0; i < 8; i++) { a += s1[i]; q += s2[i]; }
        s1[0] = a; s2[0] = q;
    }
    __syncthreads();
    float mean = s1[0] / C;
    float var  = s2[0] / C - mean * mean;
    float inv  = rsqrtf(var + 1e-6f);
    for (int c = tid; c < C; c += 256) {
        float v = (xr[c] - mean) * inv * s[c] + b[c];
        v = (1.f + scale[c]) * v + shift[c];
        out[(size_t)row * C + c] = f2bf(v);
    }
}

// ---------------------------------------------------------------------------
// Kernel 3: generic GEMM  C(MxN) = A_bf16(MxK) @ B_bf16(KxN) + bias
//   EPI=0 : outF = acc+bias
//   EPI=1 : outH = bf16(gelu(acc+bias))
//   EPI=2 : outF = res + gate[col]*(acc+bias)
// grid (N/128, M/128) x 256; 8 waves, each 32x64; BK=32; double-buffered LDS
// staged with GLOBAL_LOAD_ASYNC_TO_LDS_B128 when available.
// ---------------------------------------------------------------------------
#define ASTR 40   // A LDS row stride (halves)
#define BSTR 136  // B LDS row stride (halves)

template <int EPI>
__global__ __launch_bounds__(256) void gemm_bf16_kernel(
    const unsigned short* __restrict__ A, const unsigned short* __restrict__ B,
    const float* __restrict__ bias, const float* __restrict__ gate,
    const float* __restrict__ res, float* __restrict__ outF,
    unsigned short* __restrict__ outH, int M, int N, int K)
{
    __shared__ unsigned short As[2][128 * ASTR];
    __shared__ unsigned short Bs[2][32 * BSTR];
    int tid = threadIdx.x, lane = tid & 31, wid = tid >> 5;
    int wm = wid & 3, wn = wid >> 2;
    int bm = blockIdx.y, bn = blockIdx.x;
    int kh = ((lane >> 4) & 1) * 8;

    v8f acc[2][4];
    #pragma unroll
    for (int i = 0; i < 2; i++)
        #pragma unroll
        for (int j = 0; j < 4; j++)
            #pragma unroll
            for (int e = 0; e < 8; e++) acc[i][j][e] = 0.f;

    // tile staging: 4 async 16B chunks per thread (2 for A, 2 for B)
    auto issue_tile = [&](int buf, int kt) {
        #pragma unroll
        for (int it = 0; it < 2; it++) {
            int q = tid + it * 256;           // A: 512 chunks (128 rows x 4)
            int r = q >> 2, c8 = (q & 3) * 8;
            async_copy16(&As[buf][r * ASTR + c8],
                         A + (size_t)(bm * 128 + r) * K + kt + c8);
        }
        #pragma unroll
        for (int it = 0; it < 2; it++) {
            int q = tid + it * 256;           // B: 512 chunks (32 rows x 16)
            int k = q >> 4, c8 = (q & 15) * 8;
            async_copy16(&Bs[buf][k * BSTR + c8],
                         B + (size_t)(kt + k) * N + bn * 128 + c8);
        }
    };

    int nk = K / 32;
    issue_tile(0, 0);
    for (int t = 0; t < nk; t++) {
        int buf = t & 1;
        if (t + 1 < nk) {
            issue_tile(buf ^ 1, (t + 1) * 32);
            WAIT_ASYNC(4);
        } else {
            WAIT_ASYNC(0);
        }
        __syncthreads();

        FragB af[2];
        #pragma unroll
        for (int mi = 0; mi < 2; mi++) {
            const unsigned short* p = &As[buf][(wm * 32 + mi * 16 + (lane & 15)) * ASTR];
            #pragma unroll
            for (int j = 0; j < 4; j++) af[mi].u[j]     = *(const unsigned*)(p + kh + 2 * j);
            #pragma unroll
            for (int j = 0; j < 4; j++) af[mi].u[4 + j] = *(const unsigned*)(p + 16 + kh + 2 * j);
        }
        #pragma unroll
        for (int ni = 0; ni < 4; ni++) {
            FragB bfr;
            const unsigned short* p = &Bs[buf][wn * 64 + ni * 16 + (lane & 15)];
            #pragma unroll
            for (int v = 0; v < 8; v++) { int k0 = kh + 2 * v + 2 * (v & 4); bfr.u[v] = (unsigned)p[k0 * BSTR] | ((unsigned)p[(k0 + 1) * BSTR] << 16); }
            #pragma unroll
            for (int mi = 0; mi < 2; mi++)
                acc[mi][ni] = wmma_bf16(af[mi], bfr, acc[mi][ni]);
        }
        __syncthreads();
    }

    int rbase = bm * 128 + wm * 32 + 8 * ((lane >> 4) & 1);
    int cbase = bn * 128 + wn * 64 + (lane & 15);
    #pragma unroll
    for (int mi = 0; mi < 2; mi++) {
        #pragma unroll
        for (int ni = 0; ni < 4; ni++) {
            int col = cbase + ni * 16;
            float bv = bias ? bias[col] : 0.f;
            #pragma unroll
            for (int e = 0; e < 8; e++) {
                int row = rbase + mi * 16 + e;
                float val = acc[mi][ni][e] + bv;
                size_t idx = (size_t)row * N + col;
                if (EPI == 0) {
                    outF[idx] = val;
                } else if (EPI == 1) {
                    float x = val;
                    float g = 0.5f * x *
                        (1.f + tanhf(0.7978845608028654f * (x + 0.044715f * x * x * x)));
                    outH[idx] = f2bf(g);
                } else {
                    outF[idx] = res[idx] + gate[col] * val;
                }
            }
        }
    }
}

// ---------------------------------------------------------------------------
// Kernel 4: qkv post-process: RMS-norm q/k, RoPE, pack bf16.
// Q,V: [h][t][d];  K: TRANSPOSED [h][d][t] so flash can bulk-copy tiles.
// grid (2560, 16) x 128
// ---------------------------------------------------------------------------
__global__ __launch_bounds__(128) void qkv_post_kernel(
    const float* __restrict__ yTxt, const float* __restrict__ yImg,
    const float* __restrict__ qnT, const float* __restrict__ knT,
    const float* __restrict__ qnI, const float* __restrict__ knI,
    const float* __restrict__ pe,
    unsigned short* __restrict__ Qp, unsigned short* __restrict__ Kt,
    unsigned short* __restrict__ Vp)
{
    int T = blockIdx.x, hh = blockIdx.y, d = threadIdx.x;
    const float* y; const float* qn; const float* kn; int t;
    if (T < LT) { y = yTxt; qn = qnT; kn = knT; t = T; }
    else        { y = yImg; qn = qnI; kn = knI; t = T - LT; }
    size_t base = (size_t)t * QKVD + hh * HD + d;
    float qv = y[base], kv = y[base + HS], vv = y[base + 2 * HS];

    __shared__ float red[128];
    __shared__ float qs[128], ks[128];
    red[d] = qv * qv; __syncthreads();
    for (int st = 64; st > 0; st >>= 1) { if (d < st) red[d] += red[d + st]; __syncthreads(); }
    float qss = red[0]; __syncthreads();
    red[d] = kv * kv; __syncthreads();
    for (int st = 64; st > 0; st >>= 1) { if (d < st) red[d] += red[d + st]; __syncthreads(); }
    float kss = red[0]; __syncthreads();

    float qn_ = qv * rsqrtf(qss / (float)HD + 1e-6f) * qn[d];
    float kn_ = kv * rsqrtf(kss / (float)HD + 1e-6f) * kn[d];
    qs[d] = qn_; ks[d] = kn_;
    __syncthreads();

    int i = d >> 1, a = d & 1;
    const float* pb = pe + (size_t)T * 256 + i * 4 + a * 2;  // (...,2560,64,2,2)
    float qo = pb[0] * qs[2 * i] + pb[1] * qs[2 * i + 1];
    float ko = pb[0] * ks[2 * i] + pb[1] * ks[2 * i + 1];

    Qp[((size_t)hh * LTOT + T) * HD + d] = f2bf(qo);
    Kt[((size_t)hh * HD + d) * LTOT + T] = f2bf(ko);
    Vp[((size_t)hh * LTOT + T) * HD + d] = f2bf(vv);
}

// ---------------------------------------------------------------------------
// Kernel 5: flash attention.  grid (20, 16) x 256 (8 waves, 16 q-rows each)
// K tile [d][key] and V tile [key][d] staged as raw async 16B copies.
// ---------------------------------------------------------------------------
#define KSTR 72   // Ks row stride (halves), rows indexed by d
#define VSTR 136  // Vs row stride (halves), rows indexed by key

__global__ __launch_bounds__(256) void flash_attn_kernel(
    const unsigned short* __restrict__ Qp, const unsigned short* __restrict__ Kt,
    const unsigned short* __restrict__ Vp, unsigned short* __restrict__ Op)
{
    __shared__ unsigned short Ks[128 * KSTR];
    __shared__ unsigned short Vs[64 * VSTR];
    __shared__ unsigned short Ps[8][16 * KSTR];

    int tid = threadIdx.x, lane = tid & 31, wid = tid >> 5;
    int qb = blockIdx.x, h = blockIdx.y;
    int kh = ((lane >> 4) & 1) * 8;
    const float scl = 0.08838834764831845f;  // 1/sqrt(128)

    // Q fragments (A layout), resident for whole kernel
    int qr = qb * 128 + wid * 16 + (lane & 15);
    const unsigned short* qrow = Qp + ((size_t)h * LTOT + qr) * HD;
    FragB qf[4];
    #pragma unroll
    for (int f = 0; f < 4; f++) {
        #pragma unroll
        for (int j = 0; j < 4; j++) qf[f].u[j]     = *(const unsigned*)(qrow + f * 32 + kh + 2 * j);
        #pragma unroll
        for (int j = 0; j < 4; j++) qf[f].u[4 + j] = *(const unsigned*)(qrow + f * 32 + 16 + kh + 2 * j);
    }

    v8f o[8];
    #pragma unroll
    for (int nf = 0; nf < 8; nf++)
        #pragma unroll
        for (int e = 0; e < 8; e++) o[nf][e] = 0.f;
    float mrow[8], lrow[8];
    #pragma unroll
    for (int r = 0; r < 8; r++) { mrow[r] = -1e30f; lrow[r] = 0.f; }

    for (int kb = 0; kb < LTOT; kb += 64) {
        // stage K tile [d=128][key=64] from transposed global K
        #pragma unroll
        for (int it = 0; it < 4; it++) {
            int q = tid + it * 256;          // 1024 chunks (128 rows x 8)
            int d = q >> 3, c8 = (q & 7) * 8;
            async_copy16(Ks + d * KSTR + c8,
                         Kt + ((size_t)h * HD + d) * LTOT + kb + c8);
        }
        // stage V tile [key=64][d=128]
        #pragma unroll
        for (int it = 0; it < 2; it++) {
            int q = tid + it * 256;          // 512 chunks (64 rows x 16)
            int key = q >> 4, c8 = (q & 15) * 8;
            async_copy16(Vs + key * VSTR + c8,
                         Vp + ((size_t)h * LTOT + kb + key) * HD + c8);
        }
        WAIT_ASYNC(0);
        __syncthreads();

        // S = Q @ K^T  (16 x 64 per wave)
        v8f s[4];
        #pragma unroll
        for (int nb = 0; nb < 4; nb++) {
            #pragma unroll
            for (int e = 0; e < 8; e++) s[nb][e] = 0.f;
            #pragma unroll
            for (int f = 0; f < 4; f++) {
                FragB kf;
                const unsigned short* p = Ks + (f * 32) * KSTR + nb * 16 + (lane & 15);
                #pragma unroll
                for (int v = 0; v < 8; v++) { int k0 = kh + 2 * v + 2 * (v & 4); kf.u[v] = (unsigned)p[k0 * KSTR] | ((unsigned)p[(k0 + 1) * KSTR] << 16); }
                s[nb] = wmma_bf16(qf[f], kf, s[nb]);
            }
        }
        // online softmax
        float bmax[8];
        #pragma unroll
        for (int nb = 0; nb < 4; nb++)
            #pragma unroll
            for (int r = 0; r < 8; r++) s[nb][r] *= scl;
        #pragma unroll
        for (int r = 0; r < 8; r++)
            bmax[r] = fmaxf(fmaxf(s[0][r], s[1][r]), fmaxf(s[2][r], s[3][r]));
        #pragma unroll
        for (int off = 1; off < 16; off <<= 1)
            #pragma unroll
            for (int r = 0; r < 8; r++)
                bmax[r] = fmaxf(bmax[r], __shfl_xor(bmax[r], off, 32));
        float alpha[8], rs[8];
        #pragma unroll
        for (int r = 0; r < 8; r++) {
            float mn = fmaxf(mrow[r], bmax[r]);
            alpha[r] = __expf(mrow[r] - mn);
            mrow[r] = mn; rs[r] = 0.f;
        }
        #pragma unroll
        for (int nb = 0; nb < 4; nb++)
            #pragma unroll
            for (int r = 0; r < 8; r++) {
                float p = __expf(s[nb][r] - mrow[r]);
                s[nb][r] = p; rs[r] += p;
            }
        #pragma unroll
        for (int off = 1; off < 16; off <<= 1)
            #pragma unroll
            for (int r = 0; r < 8; r++)
                rs[r] += __shfl_xor(rs[r], off, 32);
        #pragma unroll
        for (int r = 0; r < 8; r++) lrow[r] = lrow[r] * alpha[r] + rs[r];
        #pragma unroll
        for (int nf = 0; nf < 8; nf++)
            #pragma unroll
            for (int r = 0; r < 8; r++) o[nf][r] *= alpha[r];

        // P -> wave-private LDS (C layout -> A layout round-trip)
        #pragma unroll
        for (int nb = 0; nb < 4; nb++)
            #pragma unroll
            for (int r = 0; r < 8; r++)
                Ps[wid][(r + 8 * (lane >> 4)) * KSTR + nb * 16 + (lane & 15)] = f2bf(s[nb][r]);
        asm volatile("s_wait_dscnt 0" ::: "memory");

        FragB pf[2];
        #pragma unroll
        for (int p2 = 0; p2 < 2; p2++) {
            const unsigned short* p = Ps[wid] + (lane & 15) * KSTR + p2 * 32;
            #pragma unroll
            for (int j = 0; j < 4; j++) pf[p2].u[j]     = *(const unsigned*)(p + kh + 2 * j);
            #pragma unroll
            for (int j = 0; j < 4; j++) pf[p2].u[4 + j] = *(const unsigned*)(p + 16 + kh + 2 * j);
        }
        // O += P @ V
        #pragma unroll
        for (int nf = 0; nf < 8; nf++) {
            #pragma unroll
            for (int p2 = 0; p2 < 2; p2++) {
                FragB vf;
                const unsigned short* p = Vs + (p2 * 32) * VSTR + nf * 16 + (lane & 15);
                #pragma unroll
                for (int v = 0; v < 8; v++) { int k0 = kh + 2 * v + 2 * (v & 4); vf.u[v] = (unsigned)p[k0 * VSTR] | ((unsigned)p[(k0 + 1) * VSTR] << 16); }
                o[nf] = wmma_bf16(pf[p2], vf, o[nf]);
            }
        }
        __syncthreads();
    }

    #pragma unroll
    for (int nf = 0; nf < 8; nf++) {
        int d = nf * 16 + (lane & 15);
        #pragma unroll
        for (int r = 0; r < 8; r++) {
            int row = qb * 128 + wid * 16 + r + 8 * (lane >> 4);
            float val = o[nf][r] / lrow[r];
            Op[(size_t)row * HS + h * HD + d] = f2bf(val);
        }
    }
}

// ---------------------------------------------------------------------------
// Host-side launcher
// ---------------------------------------------------------------------------
extern "C" void kernel_launch(void* const* d_in, const int* in_sizes, int n_in,
                              void* d_out, int out_size, void* d_ws, size_t ws_size,
                              hipStream_t stream) {
    (void)in_sizes; (void)n_in; (void)out_size; (void)ws_size;

    const float* img = (const float*)d_in[0];
    const float* txt = (const float*)d_in[1];
    const float* vec = (const float*)d_in[2];
    const float* pe  = (const float*)d_in[3];
    const float* i_mod_w  = (const float*)d_in[4];
    const float* i_mod_b  = (const float*)d_in[5];
    const float* i_n1_s   = (const float*)d_in[6];
    const float* i_n1_b   = (const float*)d_in[7];
    const float* i_qkv_w  = (const float*)d_in[8];
    const float* i_qkv_b  = (const float*)d_in[9];
    const float* i_qn_s   = (const float*)d_in[10];
    const float* i_kn_s   = (const float*)d_in[11];
    const float* i_proj_w = (const float*)d_in[12];
    const float* i_proj_b = (const float*)d_in[13];
    const float* i_n2_s   = (const float*)d_in[14];
    const float* i_n2_b   = (const float*)d_in[15];
    const float* i_mlp_w1 = (const float*)d_in[16];
    const float* i_mlp_b1 = (const float*)d_in[17];
    const float* i_mlp_w2 = (const float*)d_in[18];
    const float* i_mlp_b2 = (const float*)d_in[19];
    const float* t_mod_w  = (const float*)d_in[20];
    const float* t_mod_b  = (const float*)d_in[21];
    const float* t_n1_s   = (const float*)d_in[22];
    const float* t_n1_b   = (const float*)d_in[23];
    const float* t_qkv_w  = (const float*)d_in[24];
    const float* t_qkv_b  = (const float*)d_in[25];
    const float* t_qn_s   = (const float*)d_in[26];
    const float* t_kn_s   = (const float*)d_in[27];
    const float* t_proj_w = (const float*)d_in[28];
    const float* t_proj_b = (const float*)d_in[29];
    const float* t_n2_s   = (const float*)d_in[30];
    const float* t_n2_b   = (const float*)d_in[31];
    const float* t_mlp_w1 = (const float*)d_in[32];
    const float* t_mlp_b1 = (const float*)d_in[33];
    const float* t_mlp_w2 = (const float*)d_in[34];
    const float* t_mlp_b2 = (const float*)d_in[35];

    // workspace layout (bytes)
    char* ws = (char*)d_ws;
    float*          mI    = (float*)(ws + 0);                    // 12288 f32
    float*          mT    = mI + MODD;                           // 12288 f32
    unsigned short* xmImg = (unsigned short*)(ws + 98304);       // 2048x2048 bf16
    unsigned short* xmTxt = (unsigned short*)(ws + 8487168);     // 512x2048 bf16
    float*          yImg  = (float*)(ws + 10584320);             // 2048x6144 f32 (reused as h1Img)
    float*          yTxt  = (float*)(ws + 60915968);             // 512x6144 f32 (reused as h1Txt)
    unsigned short* Qp    = (unsigned short*)(ws + 73498880);    // 16x2560x128 bf16
    unsigned short* Ktr   = (unsigned short*)(ws + 83984640);    // 16x128x2560 bf16
    unsigned short* Vp    = (unsigned short*)(ws + 94470400);
    unsigned short* attn  = (unsigned short*)(ws + 104956160);   // 2560x2048 bf16
    float*          sbImg = (float*)(ws + 115441920);            // 2048x2048 f32
    float*          sbTxt = (float*)(ws + 132219136);            // 512x2048 f32
    unsigned short* wbuf  = (unsigned short*)(ws + 136413440);   // 16.8M bf16 weight scratch
    unsigned short* h1Img = (unsigned short*)yImg;
    unsigned short* h1Txt = (unsigned short*)yTxt;

    float* outImg = (float*)d_out;
    float* outTxt = outImg + (size_t)LI * HS;

    // 1. modulation vectors
    mod_kernel<<<dim3(MODD / 256, 2), 256, 0, stream>>>(
        vec, i_mod_w, i_mod_b, t_mod_w, t_mod_b, mI, mT);

    // 2. norm1 + modulation
    ln_mod_kernel<<<LI, 256, 0, stream>>>(img, i_n1_s, i_n1_b, mI + 0 * HS, mI + 1 * HS, xmImg);
    ln_mod_kernel<<<LT, 256, 0, stream>>>(txt, t_n1_s, t_n1_b, mT + 0 * HS, mT + 1 * HS, xmTxt);

    // 3. qkv GEMMs (convert weights to bf16 once, then async-staged WMMA GEMM)
    cvt_bf16_kernel<<<HS * QKVD / 2048, 256, 0, stream>>>(i_qkv_w, wbuf);
    gemm_bf16_kernel<0><<<dim3(QKVD / 128, LI / 128), 256, 0, stream>>>(
        xmImg, wbuf, i_qkv_b, nullptr, nullptr, yImg, nullptr, LI, QKVD, HS);
    cvt_bf16_kernel<<<HS * QKVD / 2048, 256, 0, stream>>>(t_qkv_w, wbuf);
    gemm_bf16_kernel<0><<<dim3(QKVD / 128, LT / 128), 256, 0, stream>>>(
        xmTxt, wbuf, t_qkv_b, nullptr, nullptr, yTxt, nullptr, LT, QKVD, HS);

    // 4. rms-norm q/k + rope + pack (K transposed)
    qkv_post_kernel<<<dim3(LTOT, NH), 128, 0, stream>>>(
        yTxt, yImg, t_qn_s, t_kn_s, i_qn_s, i_kn_s, pe, Qp, Ktr, Vp);

    // 5. flash attention
    flash_attn_kernel<<<dim3(LTOT / 128, NH), 256, 0, stream>>>(Qp, Ktr, Vp, attn);

    // 6. out projection + gated residual
    cvt_bf16_kernel<<<HS * HS / 2048, 256, 0, stream>>>(i_proj_w, wbuf);
    gemm_bf16_kernel<2><<<dim3(HS / 128, LI / 128), 256, 0, stream>>>(
        attn + (size_t)LT * HS, wbuf, i_proj_b, mI + 2 * HS, img, sbImg, nullptr, LI, HS, HS);
    cvt_bf16_kernel<<<HS * HS / 2048, 256, 0, stream>>>(t_proj_w, wbuf);
    gemm_bf16_kernel<2><<<dim3(HS / 128, LT / 128), 256, 0, stream>>>(
        attn, wbuf, t_proj_b, mT + 2 * HS, txt, sbTxt, nullptr, LT, HS, HS);

    // 7. norm2 + modulation
    ln_mod_kernel<<<LI, 256, 0, stream>>>(sbImg, i_n2_s, i_n2_b, mI + 3 * HS, mI + 4 * HS, xmImg);
    ln_mod_kernel<<<LT, 256, 0, stream>>>(sbTxt, t_n2_s, t_n2_b, mT + 3 * HS, mT + 4 * HS, xmTxt);

    // 8. mlp1 (GELU, bf16 out)
    cvt_bf16_kernel<<<HS * MLPD / 2048, 256, 0, stream>>>(i_mlp_w1, wbuf);
    gemm_bf16_kernel<1><<<dim3(MLPD / 128, LI / 128), 256, 0, stream>>>(
        xmImg, wbuf, i_mlp_b1, nullptr, nullptr, nullptr, h1Img, LI, MLPD, HS);
    cvt_bf16_kernel<<<HS * MLPD / 2048, 256, 0, stream>>>(t_mlp_w1, wbuf);
    gemm_bf16_kernel<1><<<dim3(MLPD / 128, LT / 128), 256, 0, stream>>>(
        xmTxt, wbuf, t_mlp_b1, nullptr, nullptr, nullptr, h1Txt, LT, MLPD, HS);

    // 9. mlp2 + gated residual -> final outputs
    cvt_bf16_kernel<<<MLPD * HS / 2048, 256, 0, stream>>>(i_mlp_w2, wbuf);
    gemm_bf16_kernel<2><<<dim3(HS / 128, LI / 128), 256, 0, stream>>>(
        h1Img, wbuf, i_mlp_b2, mI + 5 * HS, sbImg, outImg, nullptr, LI, HS, MLPD);
    cvt_bf16_kernel<<<MLPD * HS / 2048, 256, 0, stream>>>(t_mlp_w2, wbuf);
    gemm_bf16_kernel<2><<<dim3(HS / 128, LT / 128), 256, 0, stream>>>(
        h1Txt, wbuf, t_mlp_b2, mT + 5 * HS, sbTxt, outTxt, nullptr, LT, HS, MLPD);
}
